// VectorQuantizer_15985868275765
// MI455X (gfx1250) — matrix operand order repaired and checked by reading, not compiled, not run on hardware
//
#include <hip/hip_runtime.h>

// ---------------- problem constants ----------------
#define B_   16
#define D_   256
#define T_   4096
#define K_   1024
#define N_   65536   // B_*T_
#define MROW 128     // rows per argmin block
#define WCT  64      // codes per LDS W tile
#define NCT  (K_ / WCT)   // 16 W tiles
#define XPAD 264     // padded LDS row stride (bf16 elems), 16B-aligned rows

typedef __attribute__((ext_vector_type(16))) __bf16 v16bf;
typedef __attribute__((ext_vector_type(8)))  __bf16 v8bf;
typedef __attribute__((ext_vector_type(8)))  float  v8f;
typedef __attribute__((ext_vector_type(4)))  float  f32x4;
typedef __attribute__((ext_vector_type(4)))  unsigned int u32x4;
typedef __attribute__((ext_vector_type(4)))  int    v4i;

#if __has_builtin(__builtin_amdgcn_global_load_async_to_lds_b128) && \
    __has_builtin(__builtin_amdgcn_s_wait_asynccnt)
#define USE_ASYNC_LDS 1
#else
#define USE_ASYNC_LDS 0
#endif

// ---- fragment loaders (per CDNA5 ISA §7.12.2 16-bit layouts) ----
// A 16x32 bf16: lane m (0-15): M=m, K={0..7,16..23}; lanes 16-31: M=m, K={8..15,24..31}
__device__ inline v16bf load_frag_a(const __bf16* p) {
  union { v16bf f; v8bf h[2]; } u;
  u.h[0] = *(const v8bf*)(p);
  u.h[1] = *(const v8bf*)(p + 16);
  return u.f;
}
// B 32x16 bf16: lane n (0-15): N=n, K={0..7,16..23}; lanes 16-31: N=n, K={8..15,24..31}
__device__ inline v16bf load_frag_b(const __bf16* p) {
  union { v16bf f; v8bf h[2]; } u;
  u.h[0] = *(const v8bf*)(p);
  u.h[1] = *(const v8bf*)(p + 16);
  return u.f;
}

// Stage one 64-row bf16 codebook tile global->LDS (async on CDNA5, sync fallback)
__device__ inline void stage_wtile(const __bf16* __restrict__ wg,
                                   __bf16* __restrict__ dst, int tid) {
  #pragma unroll
  for (int i = tid; i < WCT * (D_ / 8); i += 256) {   // 8 x 16B chunks per thread
    const int code = i >> 5, chunk = i & 31;
#if USE_ASYNC_LDS
    __builtin_amdgcn_global_load_async_to_lds_b128(
        (v4i*)(wg + code * D_ + chunk * 8),
        (v4i*)(dst + code * XPAD + chunk * 8), 0, 0);
#else
    *(u32x4*)(dst + code * XPAD + chunk * 8) =
        *(const u32x4*)(wg + code * D_ + chunk * 8);
#endif
  }
}

// ---------------- kernel 0: zero histogram ----------------
__global__ void vq_zero(float* __restrict__ counts) {
  counts[threadIdx.x] = 0.0f;   // <<<1,1024>>>
}

// ---------------- kernel 1: weight -> bf16 + exact ||w||^2 ----------------
__global__ __launch_bounds__(256) void vq_prep(const float* __restrict__ w,
                                               __bf16* __restrict__ wbf,
                                               float* __restrict__ w2) {
  __shared__ float red[8];
  const int k = blockIdx.x, t = threadIdx.x;
  const float v = w[k * D_ + t];
  wbf[k * D_ + t] = (__bf16)v;
  float s = v * v;
  #pragma unroll
  for (int off = 16; off >= 1; off >>= 1) s += __shfl_xor(s, off, 32);
  if ((t & 31) == 0) red[t >> 5] = s;
  __syncthreads();
  if (t == 0) {
    float tot = 0.f;
    #pragma unroll
    for (int i = 0; i < 8; ++i) tot += red[i];
    w2[k] = tot;
  }
}

// ---------------- kernel 2: WMMA distance + argmin ----------------
// grid = N_/MROW = 512 blocks, 256 threads (8 waves), dynamic LDS
__global__ __launch_bounds__(256) void vq_argmin(const float* __restrict__ inp,
                                                 const __bf16* __restrict__ wbf,
                                                 const float* __restrict__ w2,
                                                 int* __restrict__ idxbuf) {
  extern __shared__ char smem[];
  __bf16* xsh = (__bf16*)smem;                               // [MROW][XPAD]
  __bf16* wsh0 = (__bf16*)(smem + MROW * XPAD * 2);          // [WCT][XPAD] buf 0
  __bf16* wsh1 = wsh0 + WCT * XPAD;                          // [WCT][XPAD] buf 1

  const int tid  = threadIdx.x;
  const int lane = tid & 31;
  const int wave = tid >> 5;
  const int hf   = lane >> 4;
  const int m    = lane & 15;

  const int rowBlock = blockIdx.x * MROW;   // rows don't cross batch boundary (4096%128==0)
  const int b  = rowBlock >> 12;
  const int t0 = rowBlock & (T_ - 1);
  const float* xbase = inp + (size_t)b * (size_t)(D_ * T_) + t0;

  // Kick off tile 0 staging first so it overlaps the x-tile conversion below.
  stage_wtile(wbf, wsh0, tid);

  // Stage x tile [128 rows x 256 dims] -> LDS bf16 (coalesced 128B global segments)
  for (int i = tid; i < MROW * D_; i += 256) {
    const int d = i >> 7, r = i & (MROW - 1);
    xsh[r * XPAD + d] = (__bf16)xbase[(size_t)d * T_ + r];
  }
  __syncthreads();

  // Pre-load all 8 A-fragments for this wave's 16-row tile (reused 64x)
  const int rt = wave * 16;
  v16bf afrag[8];
  #pragma unroll
  for (int kb = 0; kb < 8; ++kb)
    afrag[kb] = load_frag_a(xsh + (rt + m) * XPAD + kb * 32 + hf * 8);

  float minv[8];
  int   mini[8];
  #pragma unroll
  for (int j = 0; j < 8; ++j) { minv[j] = __builtin_inff(); mini[j] = 0; }

  for (int ct = 0; ct < NCT; ++ct) {
    __bf16* wcur = (ct & 1) ? wsh1 : wsh0;
    // Prefetch next W tile into the other buffer (overlaps with this tile's WMMAs)
    if (ct + 1 < NCT)
      stage_wtile(wbf + (size_t)((ct + 1) * WCT) * D_, (ct & 1) ? wsh0 : wsh1, tid);
#if USE_ASYNC_LDS
    // Each wave has <=8 newest (next-tile) asyncs outstanding; <=8 => tile ct resident.
    if (ct + 1 < NCT) __builtin_amdgcn_s_wait_asynccnt(8);
    else              __builtin_amdgcn_s_wait_asynccnt(0);
#endif
    __syncthreads();   // tile ct visible to all waves

    #pragma unroll
    for (int cs = 0; cs < WCT / 16; ++cs) {
      // Batch all 8 B fragments first so ds_load_b128s issue back-to-back,
      // then run the 8 WMMAs without per-WMMA dscnt waits.
      v16bf bfr[8];
      #pragma unroll
      for (int kb = 0; kb < 8; ++kb)
        bfr[kb] = load_frag_b(wcur + (cs * 16 + m) * XPAD + kb * 32 + hf * 8);

      v8f c = {};
      #pragma unroll
      for (int kb = 0; kb < 8; ++kb)
        c = __builtin_amdgcn_wmma_f32_16x16x32_bf16(
                false, afrag[kb], false, bfr[kb], (short)0, c, false, false);

      const int code = ct * WCT + cs * 16 + m;
      const float w2v = w2[code];
      // argmin is invariant to +||x||^2, so rank by ||w||^2 - 2 x.w
      #pragma unroll
      for (int j = 0; j < 8; ++j) {
        const float dist = __builtin_fmaf(-2.0f, c[j], w2v);
        if (dist < minv[j]) { minv[j] = dist; mini[j] = code; }
      }
    }
    __syncthreads();   // all waves done reading wcur before it is re-staged (ct+2)
  }

  // reduce (min, first-index) across the 16 lanes sharing each row
  #pragma unroll
  for (int off = 8; off >= 1; off >>= 1) {
    #pragma unroll
    for (int j = 0; j < 8; ++j) {
      const float om = __shfl_xor(minv[j], off, 32);
      const int   oi = __shfl_xor(mini[j], off, 32);
      if (om < minv[j] || (om == minv[j] && oi < mini[j])) { minv[j] = om; mini[j] = oi; }
    }
  }
  if (m == 0) {
    const int base = rowBlock + rt + hf * 8;   // C layout: vgpr j -> row j (+8 for upper half)
    #pragma unroll
    for (int j = 0; j < 8; ++j) idxbuf[base + j] = mini[j];
  }
}

// ---------------- kernel 3: gather quantized rows + histogram + idx out ----------------
// grid = N_/4 blocks, 256 threads; each 64-thread group emits one 1KB row (float4 coalesced)
__global__ __launch_bounds__(256) void vq_write(const float* __restrict__ w,
                                                const int* __restrict__ idxbuf,
                                                float* __restrict__ outq,
                                                float* __restrict__ outidx,
                                                float* __restrict__ counts) {
  const int n = blockIdx.x * 4 + (threadIdx.x >> 6);
  const int t = threadIdx.x & 63;
  const int k = idxbuf[n];
  const f32x4* src = (const f32x4*)(w + (size_t)k * D_);
  f32x4* dst = (f32x4*)(outq + (size_t)n * D_);
  dst[t] = src[t];
  if (t == 0) {
    outidx[n] = (float)k;
    atomicAdd(&counts[k], 1.0f);   // exact integer adds -> order-independent
  }
}

// ---------------- kernel 4: deterministic loss partials ----------------
// grid = 512 blocks, 256 threads; same coalesced x access pattern as vq_argmin
__global__ __launch_bounds__(256) void vq_loss(const float* __restrict__ inp,
                                               const float* __restrict__ w,
                                               const int* __restrict__ idxbuf,
                                               float* __restrict__ partial) {
  __shared__ int   idxs[MROW];
  __shared__ float red[8];
  const int tid = threadIdx.x;
  const int rowBlock = blockIdx.x * MROW;
  const int b = rowBlock >> 12, t0 = rowBlock & (T_ - 1);
  if (tid < MROW) idxs[tid] = idxbuf[rowBlock + tid];
  __syncthreads();
  const float* xbase = inp + (size_t)b * (size_t)(D_ * T_) + t0;
  float acc = 0.f;
  for (int i = tid; i < MROW * D_; i += 256) {
    const int d = i >> 7, r = i & (MROW - 1);
    const float xv = xbase[(size_t)d * T_ + r];
    const float qv = w[(size_t)idxs[r] * D_ + d];
    const float df = qv - xv;
    acc = __builtin_fmaf(df, df, acc);
  }
  #pragma unroll
  for (int off = 16; off >= 1; off >>= 1) acc += __shfl_xor(acc, off, 32);
  if ((tid & 31) == 0) red[tid >> 5] = acc;
  __syncthreads();
  if (tid == 0) {
    float s = 0.f;
    #pragma unroll
    for (int i = 0; i < 8; ++i) s += red[i];
    partial[blockIdx.x] = s;   // deterministic: no float atomics
  }
}

// ---------------- kernel 5: scalars (loss, perplexity) ----------------
__global__ __launch_bounds__(1024) void vq_final(const float* __restrict__ counts,
                                                 const float* __restrict__ partial,
                                                 float* __restrict__ out_loss,
                                                 float* __restrict__ out_perp) {
  __shared__ float redE[32], redL[32];
  const int tid = threadIdx.x;
  const float p = counts[tid] * (1.0f / (float)N_);
  float e = -p * logf(p + 1e-10f);
  float l = (tid < 512) ? partial[tid] : 0.0f;
  #pragma unroll
  for (int off = 16; off >= 1; off >>= 1) {
    e += __shfl_xor(e, off, 32);
    l += __shfl_xor(l, off, 32);
  }
  if ((tid & 31) == 0) { redE[tid >> 5] = e; redL[tid >> 5] = l; }
  __syncthreads();
  if (tid == 0) {
    float se = 0.f, sl = 0.f;
    #pragma unroll
    for (int i = 0; i < 32; ++i) { se += redE[i]; sl += redL[i]; }
    *out_perp = expf(se);
    // loss = q_latent + 0.25*e_latent = 1.25 * mean(diff^2)
    *out_loss = sl * (1.25f / ((float)N_ * (float)D_));
  }
}

// ---------------- launcher ----------------
extern "C" void kernel_launch(void* const* d_in, const int* in_sizes, int n_in,
                              void* d_out, int out_size, void* d_ws, size_t ws_size,
                              hipStream_t stream) {
  const float* inp = (const float*)d_in[0];   // [16, 256, 4096] f32
  const float* w   = (const float*)d_in[1];   // [1024, 256] f32
  float* out = (float*)d_out;

  // workspace carve-up (all 16B-aligned)
  char* ws = (char*)d_ws;
  int*    idxbuf  = (int*)ws;                                   // 256 KB
  __bf16* wbf     = (__bf16*)(ws + 262144);                     // 512 KB
  float*  w2      = (float*)(ws + 262144 + 524288);             // 4 KB
  float*  counts  = w2 + K_;                                    // 4 KB
  float*  partial = counts + K_;                                // 2 KB

  // output carve-up: quantized_st [N,D] | loss | perplexity | indices [N]
  float* outq     = out;
  float* out_loss = out + (size_t)N_ * D_;
  float* out_perp = out_loss + 1;
  float* outidx   = out_perp + 1;

  vq_zero<<<1, K_, 0, stream>>>(counts);
  vq_prep<<<K_, 256, 0, stream>>>(w, wbf, w2);

  const size_t shbytes = (size_t)(MROW * XPAD + 2 * WCT * XPAD) * sizeof(__bf16); // ~132 KB
  vq_argmin<<<N_ / MROW, 256, shbytes, stream>>>(inp, wbf, w2, idxbuf);

  vq_write<<<N_ / 4, 256, 0, stream>>>(w, idxbuf, outq, outidx, counts);
  vq_loss<<<N_ / MROW, 256, 0, stream>>>(inp, w, idxbuf, partial);
  vq_final<<<1, 1024, 0, stream>>>(counts, partial, out_loss, out_perp);
}
